// S4Stack_66881230733392
// MI455X (gfx1250) — matrix operand order repaired and checked by reading, not compiled, not run on hardware
//
#include <hip/hip_runtime.h>
#include <hip/hip_bf16.h>
#include <math.h>

// ---------------------------------------------------------------------------
// S4 stack for MI455X (gfx1250, wave32, WMMA + async-LDS staging).
//   - Bidirectional S4 conv as per-channel Toeplitz GEMM on
//     v_wmma_f32_16x16x32_f16; Toeplitz tiles synthesized from a reversed
//     2KB LDS kernel table (ascending-stride fragment loads -> clean b128).
//   - Output projection: 16384x768 x 768x768 WMMA GEMM, double-buffered LDS
//     operands staged with global_load_async_to_lds_b128 (ASYNCcnt pipeline).
//   - LN / gelu / kernel-synthesis as fused VALU kernels.
// ---------------------------------------------------------------------------

#define Bq 32
#define Lq 512
#define Hq 768
#define Nq 64
#define NLq 4

typedef __attribute__((ext_vector_type(16))) _Float16 v16h;
typedef __attribute__((ext_vector_type(8)))  float    v8f;

// 16B global -> LDS async copy (per-lane), tracked by ASYNCcnt.
__device__ __forceinline__ void async_copy16(unsigned int lds_addr, const _Float16* g)
{
  asm volatile("global_load_async_to_lds_b128 %0, %1, off"
               :: "v"(lds_addr), "v"(g) : "memory");
}
__device__ __forceinline__ unsigned int lds_off(const void* p)
{
  return (unsigned int)(unsigned long long)p;   // low 32b of flat = LDS offset
}

// -------------------------------------------------------------------- LN1 ---
__global__ __launch_bounds__(256) void k_ln1(
    const float* __restrict__ x, const int* __restrict__ mask,
    const float* __restrict__ w, const float* __restrict__ bia,
    float* __restrict__ u32, _Float16* __restrict__ u16)
{
  const int tok = blockIdx.x;            // b*L + l
  const int b = tok >> 9, l = tok & (Lq - 1);
  const int tid = threadIdx.x;
  const float* xr = x + (size_t)tok * Hq;
  float v[3], s1 = 0.f, s2 = 0.f;
#pragma unroll
  for (int i = 0; i < 3; ++i) { v[i] = xr[tid + i*256]; s1 += v[i]; s2 += v[i]*v[i]; }
  __shared__ float r1[256], r2[256];
  r1[tid] = s1; r2[tid] = s2; __syncthreads();
  for (int off = 128; off > 0; off >>= 1) {
    if (tid < off) { r1[tid] += r1[tid+off]; r2[tid] += r2[tid+off]; }
    __syncthreads();
  }
  const float mu  = r1[0] * (1.f/Hq);
  const float var = r2[0] * (1.f/Hq) - mu*mu;
  const float rs  = rsqrtf(var + 1e-5f);
  const float mf  = (float)mask[tok];
#pragma unroll
  for (int i = 0; i < 3; ++i) {
    const int hh = tid + i*256;
    const float uv = ((v[i]-mu)*rs*w[hh] + bia[hh]) * mf;
    const size_t oi = ((size_t)b*Hq + hh)*Lq + l;   // transposed [B,H,L]
    u32[oi] = uv; u16[oi] = (_Float16)uv;
  }
}

// -------------------------------------------------- S4 kernel synthesis  ---
__global__ __launch_bounds__(256) void k_s4kernel(
    const float* __restrict__ log_dt, const float* __restrict__ A_re,
    const float* __restrict__ A_im,  const float* __restrict__ C_re,
    const float* __restrict__ C_im,
    _Float16* __restrict__ K0, _Float16* __restrict__ K1)   // [H,L]
{
  __shared__ float sC0r[Nq], sC0i[Nq], sC1r[Nq], sC1i[Nq], sEr[Nq], sEi[Nq];
  const int h = blockIdx.x, tid = threadIdx.x;
  if (tid < Nq) {
    const int n = tid;
    const float dt = __expf(log_dt[h]);
    const float ar = -__expf(A_re[h*Nq + n]);
    const float ai = A_im[h*Nq + n];
    const float dr = dt*ar, di = dt*ai;
    float s, c; __sincosf(di, &s, &c);
    const float e = __expf(dr);
    const float numr = e*c - 1.f, numi = e*s;
    const float inv = 1.f / (ar*ar + ai*ai);
    const float wr = (numr*ar + numi*ai)*inv;
    const float wi = (numi*ar - numr*ai)*inv;
    const float c0r = C_re[h*Nq + n],         c0i = C_im[h*Nq + n];
    const float c1r = C_re[Hq*Nq + h*Nq + n], c1i = C_im[Hq*Nq + h*Nq + n];
    sC0r[n] = c0r*wr - c0i*wi;  sC0i[n] = c0r*wi + c0i*wr;
    sC1r[n] = c1r*wr - c1i*wi;  sC1i[n] = c1r*wi + c1i*wr;
    sEr[n] = dr; sEi[n] = di;
  }
  __syncthreads();
  for (int l = tid; l < Lq; l += 256) {
    float k0 = 0.f, k1 = 0.f;
    const float fl = (float)l;
    for (int n = 0; n < Nq; ++n) {
      const float er = __expf(sEr[n]*fl);
      float s, c; __sincosf(sEi[n]*fl, &s, &c);
      const float vr = er*c, vi = er*s;
      k0 += sC0r[n]*vr - sC0i[n]*vi;
      k1 += sC1r[n]*vr - sC1i[n]*vi;
    }
    K0[(size_t)h*Lq + l] = (_Float16)(2.f*k0);
    K1[(size_t)h*Lq + l] = (_Float16)(2.f*k1);
  }
}

// ------------------------------------------ Wout f32 -> f16 (same layout) ---
// Kept row-major [o][h]; GEMM stages B tiles n-major so no transpose needed.
__global__ __launch_bounds__(256) void k_woutT(
    const float* __restrict__ Wout, _Float16* __restrict__ wt)
{
  const int i = blockIdx.x*256 + threadIdx.x;
  if (i >= Hq*Hq) return;
  wt[i] = (_Float16)Wout[i];
}

// -------------------------------------- bidirectional conv as Toeplitz GEMM
// y[b,l] = sum_{j<=l} K0[l-j] u[b,j] + sum_{j>l} K1[j-l-1] u[b,j]
// Per h: Y[32x512] = U[32x512] x T[512x512]; T[j,l] read from reversed table
// sMr[512 - (l-j)]  (ascending in j -> contiguous b128 fragment loads).
__global__ __launch_bounds__(256) void k_conv(
    const _Float16* __restrict__ u16,   // [B,H,L]
    const _Float16* __restrict__ K0, const _Float16* __restrict__ K1,
    float* __restrict__ yout)           // [B,H,L]
{
  __shared__ _Float16 sU[Bq*Lq];        // 32 KB
  __shared__ _Float16 sMr[1024];        // reversed merged kernel table
  const int h = blockIdx.x, nblk = blockIdx.y, tid = threadIdx.x;

  for (int t = tid; t < (Bq*Lq)/8; t += 256) {   // async 16B fills (8/thread)
    const int b = t >> 6, q = t & 63;
    async_copy16(lds_off(sU + (size_t)b*Lq + q*8),
                 u16 + ((size_t)b*Hq + h)*Lq + q*8);
  }
  for (int t = tid; t < 1024; t += 256) {        // sMr[i]: i=512-d
    _Float16 v;
    if (t == 0)        v = (_Float16)0.f;                 // unused pad
    else if (t <= 512) v = K0[(size_t)h*Lq + (512 - t)];  // d>=0 : K0[d]
    else               v = K1[(size_t)h*Lq + (t - 513)];  // d<0  : K1[-1-d]
    sMr[t] = v;
  }
  asm volatile("s_wait_asynccnt 0x0" ::: "memory");
  __syncthreads();

  const int lane = tid & 31, wave = tid >> 5;
  const int col = nblk*128 + wave*16 + (lane & 15);    // output l
  v8f acc0, acc1;
#pragma unroll
  for (int r = 0; r < 8; ++r) { acc0[r] = 0.f; acc1[r] = 0.f; }

  for (int ks = 0; ks < Lq/32; ++ks) {
    const int k = ks*32;
    v16h bf, a0, a1;
    const int kb = k + ((lane & 16) ? 16 : 0);         // B: lanes 16-31 -> K+16
    const int mbase = 512 - col + kb;                  // ascending in idx
#pragma unroll
    for (int idx = 0; idx < 16; ++idx)
      bf[idx] = sMr[mbase + idx];
    const int abase = k + ((lane & 16) ? 16 : 0);      // A: lanes 16-31 -> K+16
    const int r0 = lane & 15;
#pragma unroll
    for (int idx = 0; idx < 16; ++idx) {
      const int kl = abase + idx;
      a0[idx] = sU[r0*Lq + kl];
      a1[idx] = sU[(r0 + 16)*Lq + kl];
    }
    acc0 = __builtin_amdgcn_wmma_f32_16x16x32_f16(false, a0, false, bf, (short)0, acc0, false, false);
    acc1 = __builtin_amdgcn_wmma_f32_16x16x32_f16(false, a1, false, bf, (short)0, acc1, false, false);
  }

#pragma unroll
  for (int r = 0; r < 8; ++r) {
    const int b0 = r + ((lane >> 4) << 3);
    yout[((size_t)b0*Hq + h)*Lq + col]      = acc0[r];
    yout[((size_t)(b0+16)*Hq + h)*Lq + col] = acc1[r];
  }
}

// ------------------------------------------------- y + D*u, gelu, f16+T ---
__global__ __launch_bounds__(256) void k_postconv(
    const float* __restrict__ yconv, const float* __restrict__ u32,
    const float* __restrict__ Dp, _Float16* __restrict__ act)  // act: [B,L,H]
{
  const size_t i = (size_t)blockIdx.x*256 + threadIdx.x;
  if (i >= (size_t)Bq*Lq*Hq) return;
  const int hh = (int)(i % Hq);
  const size_t t = i / Hq;
  const int l = (int)(t % Lq), b = (int)(t / Lq);
  const size_t si = ((size_t)b*Hq + hh)*Lq + l;
  const float yv = yconv[si] + Dp[hh]*u32[si];
  const float g = 0.5f*yv*(1.f + tanhf(0.7978845608028654f*(yv + 0.044715f*yv*yv*yv)));
  act[i] = (_Float16)g;
}

// ------------------------------------------ output GEMM: [BL,H] x [H,H]  ---
// out[t,o] = sum_h act[t,h]*Wout[o,h] + bout[o].
// A staged [128 m][32 k], B staged [128 n][32 k] (n-major == Wout row-major),
// both via async-LDS double buffering.
__global__ __launch_bounds__(256) void k_gemm(
    const _Float16* __restrict__ A,    // [16384, 768] act
    const _Float16* __restrict__ Bm,   // [768 o][768 h] Wout (f16)
    const float* __restrict__ bout, float* __restrict__ out)
{
  __shared__ _Float16 sA[2][128*32];   // 2 x 8 KB
  __shared__ _Float16 sB[2][128*32];   // 2 x 8 KB
  const int bm = blockIdx.x, bn = blockIdx.y;
  const int tid = threadIdx.x, wave = tid >> 5, lane = tid & 31;
  const int wm = wave >> 2, wn = wave & 3;   // 2x4 wave grid, wave tile 64x32

  v8f acc[4][2];
#pragma unroll
  for (int i = 0; i < 4; ++i)
#pragma unroll
    for (int j = 0; j < 2; ++j)
#pragma unroll
      for (int r = 0; r < 8; ++r) acc[i][j][r] = 0.f;

  // stage one 32-deep K chunk into buffer `buf` (4 async ops per wave)
  auto issue = [&](int kk, int buf) {
    const int kc = kk*32;
    for (int t = tid; t < 512; t += 256) {
      const int row = t >> 2, q = t & 3;
      async_copy16(lds_off(&sA[buf][row*32 + q*8]),
                   A  + ((size_t)(bm*128 + row))*Hq + kc + q*8);
      async_copy16(lds_off(&sB[buf][row*32 + q*8]),
                   Bm + ((size_t)(bn*128 + row))*Hq + kc + q*8);
    }
  };

  issue(0, 0);
  for (int kk = 0; kk < Hq/32; ++kk) {
    const int cur = kk & 1;
    const bool more = (kk + 1) < Hq/32;
    if (more) issue(kk + 1, cur ^ 1);
    if (more) { asm volatile("s_wait_asynccnt 0x4" ::: "memory"); }  // chunk kk landed
    else      { asm volatile("s_wait_asynccnt 0x0" ::: "memory"); }
    __syncthreads();

    v16h af[4], bf[2];
    const int abase = (lane & 16) ? 16 : 0;
#pragma unroll
    for (int i = 0; i < 4; ++i) {
      const int row = wm*64 + i*16 + (lane & 15);
#pragma unroll
      for (int idx = 0; idx < 16; ++idx) {
        const int kl = abase + idx;
        af[i][idx] = sA[cur][row*32 + kl];
      }
    }
    const int kb = (lane & 16) ? 16 : 0;
#pragma unroll
    for (int j = 0; j < 2; ++j) {
      const int cl = wn*32 + j*16 + (lane & 15);
#pragma unroll
      for (int idx = 0; idx < 16; ++idx)
        bf[j][idx] = sB[cur][cl*32 + kb + idx];
    }
#pragma unroll
    for (int i = 0; i < 4; ++i)
#pragma unroll
      for (int j = 0; j < 2; ++j)
        acc[i][j] = __builtin_amdgcn_wmma_f32_16x16x32_f16(false, af[i], false, bf[j], (short)0, acc[i][j], false, false);
    __syncthreads();   // protect buffer before next prefetch overwrites it
  }

#pragma unroll
  for (int i = 0; i < 4; ++i) {
    const int mrow = bm*128 + wm*64 + i*16 + ((lane >> 4) << 3);
#pragma unroll
    for (int j = 0; j < 2; ++j) {
      const int ncol = bn*128 + wn*32 + j*16 + (lane & 15);
      const float bo = bout[ncol];
#pragma unroll
      for (int r = 0; r < 8; ++r)
        out[(size_t)(mrow + r)*Hq + ncol] = acc[i][j][r] + bo;
    }
  }
}

// ---------------------------------------------------- LN2 + residual add ---
__global__ __launch_bounds__(256) void k_ln2res(
    const float* __restrict__ xin, const float* __restrict__ g,
    const float* __restrict__ w, const float* __restrict__ bia,
    float* __restrict__ xout)
{
  const int tok = blockIdx.x, tid = threadIdx.x;
  const float* gr = g   + (size_t)tok*Hq;
  const float* xr = xin + (size_t)tok*Hq;
  float v[3], s1 = 0.f, s2 = 0.f;
#pragma unroll
  for (int i = 0; i < 3; ++i) { v[i] = gr[tid + i*256]; s1 += v[i]; s2 += v[i]*v[i]; }
  __shared__ float r1[256], r2[256];
  r1[tid] = s1; r2[tid] = s2; __syncthreads();
  for (int off = 128; off > 0; off >>= 1) {
    if (tid < off) { r1[tid] += r1[tid+off]; r2[tid] += r2[tid+off]; }
    __syncthreads();
  }
  const float mu  = r1[0] * (1.f/Hq);
  const float var = r2[0] * (1.f/Hq) - mu*mu;
  const float rs  = rsqrtf(var + 1e-5f);
#pragma unroll
  for (int i = 0; i < 3; ++i) {
    const int hh = tid + i*256;
    xout[(size_t)tok*Hq + hh] = xr[hh] + ((v[i]-mu)*rs*w[hh] + bia[hh]);  // GAIN=1
  }
}

// --------------------------------------------------------------- launch  ---
extern "C" void kernel_launch(void* const* d_in, const int* in_sizes, int n_in,
                              void* d_out, int out_size, void* d_ws, size_t ws_size,
                              hipStream_t stream)
{
  const float* x     = (const float*)d_in[0];
  const int*   mask  = (const int*)d_in[1];
  const float* ln1w  = (const float*)d_in[2];
  const float* ln1b  = (const float*)d_in[3];
  const float* logdt = (const float*)d_in[4];
  const float* Are   = (const float*)d_in[5];
  const float* Aim   = (const float*)d_in[6];
  const float* Cre   = (const float*)d_in[7];
  const float* Cim   = (const float*)d_in[8];
  const float* Dp    = (const float*)d_in[9];
  const float* Wout  = (const float*)d_in[10];
  const float* bout  = (const float*)d_in[11];
  const float* ln2w  = (const float*)d_in[12];
  const float* ln2b  = (const float*)d_in[13];
  float* out = (float*)d_out;

  const size_t TOK = (size_t)Bq*Lq;      // 16384
  const size_t ELT = TOK*Hq;             // 12.58M
  char* ws = (char*)d_ws;
  float*    x_cur = (float*)ws;            ws += ELT*4;
  float*    u32   = (float*)ws;            ws += ELT*4;           // reused: GEMM out
  _Float16* u16   = (_Float16*)ws;         ws += ELT*2;           // reused: act
  float*    yconv = (float*)ws;            ws += ELT*4;
  _Float16* K0    = (_Float16*)ws;         ws += (size_t)Hq*Lq*2;
  _Float16* K1    = (_Float16*)ws;         ws += (size_t)Hq*Lq*2;
  _Float16* wf16  = (_Float16*)ws;         ws += (size_t)Hq*Hq*2;
  // total workspace: ~171 MB

  for (int i = 0; i < NLq; ++i) {
    const float* xin = (i == 0) ? x : x_cur;
    float* xout = (i == NLq-1) ? out : x_cur;

    k_ln1<<<dim3((unsigned)TOK), 256, 0, stream>>>(xin, mask, ln1w + i*Hq, ln1b + i*Hq, u32, u16);
    k_s4kernel<<<dim3(Hq), 256, 0, stream>>>(logdt + i*Hq,
                                             Are + (size_t)i*Hq*Nq, Aim + (size_t)i*Hq*Nq,
                                             Cre + (size_t)i*2*Hq*Nq, Cim + (size_t)i*2*Hq*Nq,
                                             K0, K1);
    k_woutT<<<dim3((Hq*Hq + 255)/256), 256, 0, stream>>>(Wout + (size_t)i*Hq*Hq, wf16);
    k_conv<<<dim3(Hq, Lq/128), 256, 0, stream>>>(u16, K0, K1, yconv);
    k_postconv<<<dim3((unsigned)((ELT + 255)/256)), 256, 0, stream>>>(yconv, u32, Dp + i*Hq, u16);
    k_gemm<<<dim3((unsigned)(TOK/128), Hq/128), 256, 0, stream>>>(u16, wf16, bout + i*Hq, u32);
    k_ln2res<<<dim3((unsigned)TOK), 256, 0, stream>>>(xin, u32, ln2w + i*Hq, ln2b + i*Hq, xout);
  }
}
